// TrackFuse_25718264169217
// MI455X (gfx1250) — matrix operand order repaired and checked by reading, not compiled, not run on hardware
//
#include <hip/hip_runtime.h>
#include <cmath>

// ---------------------------------------------------------------------------
// Types for CDNA5 WMMA (wave32): v_wmma_f32_16x16x32_f16
// ---------------------------------------------------------------------------
typedef __attribute__((ext_vector_type(16))) _Float16 v16h;
typedef __attribute__((ext_vector_type(8)))  float    v8f;

union HFrag { v16h v; unsigned u[8]; };

static __device__ __forceinline__ unsigned short f2h(float x) {
    union { _Float16 h; unsigned short u; } c;
    c.h = (_Float16)x;
    return c.u;
}

#define NORMC  (-7.62461899f)   // -log(2048)
#define LOGNC  (6.93147181f)    //  log(1024)
#define BIGSPLIT (1 << 30)

// ---------------------------------------------------------------------------
// Generic WMMA GEMM:  Y[co][j] = epilogue( alpha * sum_k W[co][k] * X[k][j] )
//   W[co][k] at W + co*wrm + k*wrk          (arbitrary strides)
//   X[k][j]  at X + k*xrk + j   (j contiguous); rows k>=ksplit come from X2
//   Y[co][j] at Y + co*yrm + j
// Epilogue: (+bias[co]) -> (*bns[co]+bnb[co]) -> relu -> (*cscale[j]) -> (+resid[co][j])
// Block: 128 threads (4 waves). Tile: 16 (co) x 64 (j). K chunked by 32.
// Grid: (Ncol/64, Cout/16). Requires Ncol%64==0, Cout%16==0 (true for all uses).
// Load path is branch-free per chunk: full chunks take unchecked vector loads;
// the (at most one) partial chunk uses clamped addresses + selects, so no
// EXEC-divergent global loads anywhere.
// ---------------------------------------------------------------------------
__global__ void __launch_bounds__(128)
gemm_wmma_kernel(const float* __restrict__ W, int wrm, int wrk,
                 const float* __restrict__ X, int xrk,
                 const float* __restrict__ X2, int x2rk, int ksplit,
                 int Cin,
                 const float* __restrict__ bias,
                 const float* __restrict__ bns, const float* __restrict__ bnb,
                 int relu,
                 const float* __restrict__ cscale,
                 const float* __restrict__ resid, int rrm,
                 float alpha, float* __restrict__ Y, int yrm)
{
    __shared__ __align__(4) unsigned short ldsA[16 * 34];
    __shared__ __align__(4) unsigned short ldsB[64 * 34];

    const int tid  = threadIdx.x;
    const int j0   = blockIdx.x * 64;
    const int m0   = blockIdx.y * 16;
    const int wave = tid >> 5;
    const int lane = tid & 31;
    const int lrow = lane & 15;
    const int hi   = lane >> 4;

    v8f acc = {0.f, 0.f, 0.f, 0.f, 0.f, 0.f, 0.f, 0.f};

    const int arow = tid >> 3;        // 0..15  (A row)
    const int ak   = (tid & 7) * 4;   // 0..28  (A k group)
    const int bk   = tid & 31;        // 0..31  (B k row)
    const int cseg = tid >> 5;        // 0..3   (B column segment)

    for (int kc = 0; kc < Cin; kc += 32) {
        if (kc + 32 <= Cin) {
            // ================= full chunk: unchecked loads =================
            // ---- A tile (16 x 32) ----
            if (wrk == 1) {
                const float4 wv = *reinterpret_cast<const float4*>(
                    W + (size_t)(m0 + arow) * (size_t)wrm + (size_t)(kc + ak));
                ldsA[arow * 34 + ak + 0] = f2h(wv.x);
                ldsA[arow * 34 + ak + 1] = f2h(wv.y);
                ldsA[arow * 34 + ak + 2] = f2h(wv.z);
                ldsA[arow * 34 + ak + 3] = f2h(wv.w);
            } else {
                const float* wp = W + (size_t)(m0 + arow) * (size_t)wrm
                                    + (size_t)(kc + ak) * (size_t)wrk;
                float w0 = wp[0];
                float w1 = wp[(size_t)wrk];
                float w2 = wp[2 * (size_t)wrk];
                float w3 = wp[3 * (size_t)wrk];
                ldsA[arow * 34 + ak + 0] = f2h(w0);
                ldsA[arow * 34 + ak + 1] = f2h(w1);
                ldsA[arow * 34 + ak + 2] = f2h(w2);
                ldsA[arow * 34 + ak + 3] = f2h(w3);
            }
            // ---- B tile (32 x 64), column-major f16 pairs in LDS ----
            {
                const int k = kc + bk;
                const float* xp; int xr, kk;
                if (k < ksplit) { xp = X;  xr = xrk;  kk = k; }
                else            { xp = X2; xr = x2rk; kk = k - ksplit; }
                const float* base = xp + (size_t)kk * (size_t)xr + j0 + cseg * 16;
                // prefetch next chunk's row (resolve split pointer properly)
                const int k2 = k + 32;
                if (k2 < Cin) {
                    const float* xp2; int xr2, kk2;
                    if (k2 < ksplit) { xp2 = X;  xr2 = xrk;  kk2 = k2; }
                    else             { xp2 = X2; xr2 = x2rk; kk2 = k2 - ksplit; }
                    __builtin_prefetch(xp2 + (size_t)kk2 * (size_t)xr2 + j0 + cseg * 16, 0, 0);
                }
                const float4* src = reinterpret_cast<const float4*>(base);
                float4 xv0 = src[0];
                float4 xv1 = src[1];
                float4 xv2 = src[2];
                float4 xv3 = src[3];
                const int c = cseg * 16;
                ldsB[(c +  0) * 34 + bk] = f2h(xv0.x);
                ldsB[(c +  1) * 34 + bk] = f2h(xv0.y);
                ldsB[(c +  2) * 34 + bk] = f2h(xv0.z);
                ldsB[(c +  3) * 34 + bk] = f2h(xv0.w);
                ldsB[(c +  4) * 34 + bk] = f2h(xv1.x);
                ldsB[(c +  5) * 34 + bk] = f2h(xv1.y);
                ldsB[(c +  6) * 34 + bk] = f2h(xv1.z);
                ldsB[(c +  7) * 34 + bk] = f2h(xv1.w);
                ldsB[(c +  8) * 34 + bk] = f2h(xv2.x);
                ldsB[(c +  9) * 34 + bk] = f2h(xv2.y);
                ldsB[(c + 10) * 34 + bk] = f2h(xv2.z);
                ldsB[(c + 11) * 34 + bk] = f2h(xv2.w);
                ldsB[(c + 12) * 34 + bk] = f2h(xv3.x);
                ldsB[(c + 13) * 34 + bk] = f2h(xv3.y);
                ldsB[(c + 14) * 34 + bk] = f2h(xv3.z);
                ldsB[(c + 15) * 34 + bk] = f2h(xv3.w);
            }
        } else {
            // ============ tail chunk: clamped loads + selects ============
            const int kcl = Cin - 1;
            // ---- A ----
            {
                float wv[4];
                #pragma unroll
                for (int i = 0; i < 4; ++i) {
                    int k  = kc + ak + i;
                    int kq = (k < kcl) ? k : kcl;
                    wv[i] = W[(size_t)(m0 + arow) * (size_t)wrm + (size_t)kq * (size_t)wrk];
                    if (k >= Cin) wv[i] = 0.f;
                }
                #pragma unroll
                for (int i = 0; i < 4; ++i)
                    ldsA[arow * 34 + ak + i] = f2h(wv[i]);
            }
            // ---- B ----
            {
                const int k  = kc + bk;
                const int kq = (k < kcl) ? k : kcl;   // clamped, always a valid row
                const bool ok = (k < Cin);
                const float* xp; int xr, kk;
                if (kq < ksplit) { xp = X;  xr = xrk;  kk = kq; }
                else             { xp = X2; xr = x2rk; kk = kq - ksplit; }
                const float4* src = reinterpret_cast<const float4*>(
                    xp + (size_t)kk * (size_t)xr + j0 + cseg * 16);
                const int c = cseg * 16;
                #pragma unroll
                for (int c4 = 0; c4 < 4; ++c4) {
                    float4 xv = src[c4];
                    float e0 = ok ? xv.x : 0.f;
                    float e1 = ok ? xv.y : 0.f;
                    float e2 = ok ? xv.z : 0.f;
                    float e3 = ok ? xv.w : 0.f;
                    ldsB[(c + c4 * 4 + 0) * 34 + bk] = f2h(e0);
                    ldsB[(c + c4 * 4 + 1) * 34 + bk] = f2h(e1);
                    ldsB[(c + c4 * 4 + 2) * 34 + bk] = f2h(e2);
                    ldsB[(c + c4 * 4 + 3) * 34 + bk] = f2h(e3);
                }
            }
        }
        __syncthreads();

        // ---- build fragments per ISA 16-bit layouts ----
        HFrag af, bf;
        #pragma unroll
        for (int v = 0; v < 8; ++v) {
            // A 16x32: lanes0-15 row=M, VGPR v -> K = v<4 ? 2v : 16+2(v-4); lanes16-31: +8
            int kA = ((v < 4) ? (v * 2) : (16 + (v - 4) * 2)) + hi * 8;
            af.u[v] = *reinterpret_cast<const unsigned*>(&ldsA[lrow * 34 + kA]);
            // B 32x16: lanes0-15 col=N, VGPR v -> K = v<4 ? 2v : 16+2(v-4); lanes16-31: +8
            int kB = ((v < 4) ? (v * 2) : (16 + (v - 4) * 2)) + hi * 8;
            bf.u[v] = *reinterpret_cast<const unsigned*>(&ldsB[(wave * 16 + lrow) * 34 + kB]);
        }
        acc = __builtin_amdgcn_wmma_f32_16x16x32_f16(false, af.v, false, bf.v,
                                                     (short)0, acc, false, false);
        __syncthreads();
    }

    // ---- epilogue + store (C/D layout: VGPR r -> M = r + hi*8, N = lane&15) ----
    const int j = j0 + wave * 16 + lrow;
    #pragma unroll
    for (int r = 0; r < 8; ++r) {
        int co = m0 + hi * 8 + r;
        float val = acc[r] * alpha;
        if (bias)   val += bias[co];
        if (bns)    val = val * bns[co] + bnb[co];
        if (relu)   val = fmaxf(val, 0.f);
        if (cscale) val *= cscale[j];
        if (resid)  val += resid[(size_t)co * (size_t)rrm + j];
        Y[(size_t)co * (size_t)yrm + j] = val;
    }
}

// ---------------------------------------------------------------------------
// Raw-feature encoder: f = (relu((pf@W1^T+b1)*s+b)) @ W2^T + b2; L2-normalize
// rows; write transposed into desc[64][1024]. One block (64 thr) per point.
// ---------------------------------------------------------------------------
__global__ void __launch_bounds__(64)
enc_kernel(const float* __restrict__ pf,
           const float* __restrict__ w1, const float* __restrict__ b1,
           const float* __restrict__ s1, const float* __restrict__ sb1,
           const float* __restrict__ w2, const float* __restrict__ b2,
           float* __restrict__ desc)
{
    const int n = blockIdx.x;
    const int t = threadIdx.x;
    __shared__ float hid[32];
    __shared__ float red[64];
    __shared__ float nrm;

    if (t < 32) {
        float a = b1[t];
        const float* row = pf + (size_t)n * 24;
        #pragma unroll
        for (int i = 0; i < 24; ++i) a += row[i] * w1[t * 24 + i];
        a = a * s1[t] + sb1[t];
        hid[t] = fmaxf(a, 0.f);
    }
    __syncthreads();

    float o = b2[t];
    #pragma unroll
    for (int i = 0; i < 32; ++i) o += hid[i] * w2[t * 32 + i];
    red[t] = o * o;
    __syncthreads();
    for (int off = 32; off > 0; off >>= 1) {
        if (t < off) red[t] += red[t + off];
        __syncthreads();
    }
    if (t == 0) nrm = fmaxf(sqrtf(red[0]), 1e-12f);
    __syncthreads();
    desc[(size_t)t * 1024 + n] = o / nrm;
}

// desc1[d][n] = desc1_in[n][d]
__global__ void transpose_desc_kernel(const float* __restrict__ in, float* __restrict__ out)
{
    int idx = blockIdx.x * blockDim.x + threadIdx.x;
    if (idx < 64 * 1024) {
        int d = idx >> 10, n = idx & 1023;
        out[idx] = in[(size_t)n * 64 + d];
    }
}

// kx[5][1024]: rows 0..3 = kpts^T, row 4 = scores
__global__ void prep_kx_kernel(const float* __restrict__ kpts,
                               const float* __restrict__ scores,
                               float* __restrict__ kx)
{
    int idx = blockIdx.x * blockDim.x + threadIdx.x;
    if (idx < 5 * 1024) {
        int c = idx >> 10, n = idx & 1023;
        kx[idx] = (c < 4) ? kpts[(size_t)n * 4 + c] : scores[n];
    }
}

// Column softmax over S_T[1024][1024] (softmax along rows m, per column n):
// writes exp(x - max) into S, stores 1/sum per column in rinv.
__global__ void __launch_bounds__(256)
softmax_col_kernel(float* __restrict__ S, float* __restrict__ rinv)
{
    int n = blockIdx.x * blockDim.x + threadIdx.x;
    if (n >= 1024) return;
    float m = -INFINITY;
    for (int i = 0; i < 1024; ++i) m = fmaxf(m, S[(size_t)i * 1024 + n]);
    float s = 0.f;
    for (int i = 0; i < 1024; ++i) {
        float e = expf(S[(size_t)i * 1024 + n] - m);
        S[(size_t)i * 1024 + n] = e;
        s += e;
    }
    rinv[n] = 1.f / s;
}

// Fill last row/column of couplings with bin_score
__global__ void fill_bins_kernel(float* __restrict__ C, const float* __restrict__ bin)
{
    int i = blockIdx.x * blockDim.x + threadIdx.x;
    float b = bin[0];
    if (i < 1024) {
        C[(size_t)i * 1025 + 1024] = b;
        C[(size_t)1024 * 1025 + i] = b;
    } else if (i == 1024) {
        C[(size_t)1024 * 1025 + 1024] = b;
    }
}

__global__ void init_uv_kernel(float* __restrict__ u, float* __restrict__ v)
{
    int i = blockIdx.x * blockDim.x + threadIdx.x;
    if (i < 1025) u[i] = 0.f;
    else if (i < 2050) v[i - 1025] = 0.f;
}

// u[i] = log_mu[i] - LSE_j( C[i][j] + v[j] )   (block per row, online LSE reduce)
__global__ void __launch_bounds__(256)
sink_row_kernel(const float* __restrict__ C, const float* __restrict__ v,
                float* __restrict__ u)
{
    const int i = blockIdx.x;
    const int t = threadIdx.x;
    const float* row = C + (size_t)i * 1025;
    float m = -INFINITY, s = 0.f;
    for (int j = t; j < 1025; j += 256) {
        float x = row[j] + v[j];
        if (x > m) { s = s * expf(m - x) + 1.f; m = x; }
        else       { s += expf(x - m); }
    }
    __shared__ float sm[256], ss[256];
    sm[t] = m; ss[t] = s;
    __syncthreads();
    for (int o = 128; o > 0; o >>= 1) {
        if (t < o) {
            float m2 = sm[t + o], s2 = ss[t + o];
            float mm = fmaxf(sm[t], m2);
            ss[t] = ss[t] * expf(sm[t] - mm) + s2 * expf(m2 - mm);
            sm[t] = mm;
        }
        __syncthreads();
    }
    if (t == 0) {
        float lse = sm[0] + logf(ss[0]);
        float lmu = (i < 1024) ? NORMC : (LOGNC + NORMC);
        u[i] = lmu - lse;
    }
}

// v[j] = log_nu[j] - LSE_i( C[i][j] + u[i] )   (thread per column, coalesced)
__global__ void __launch_bounds__(256)
sink_col_kernel(const float* __restrict__ C, const float* __restrict__ u,
                float* __restrict__ v)
{
    int j = blockIdx.x * blockDim.x + threadIdx.x;
    if (j >= 1025) return;
    float m = -INFINITY, s = 0.f;
    for (int i = 0; i < 1025; ++i) {
        float x = C[(size_t)i * 1025 + j] + u[i];
        if (x > m) { s = s * expf(m - x) + 1.f; m = x; }
        else       { s += expf(x - m); }
    }
    float lse = m + logf(s);
    float lnu = (j < 1024) ? NORMC : (LOGNC + NORMC);
    v[j] = lnu - lse;
}

__global__ void final_out_kernel(const float* __restrict__ C,
                                 const float* __restrict__ u,
                                 const float* __restrict__ v,
                                 float* __restrict__ out)
{
    int idx = blockIdx.x * blockDim.x + threadIdx.x;
    if (idx >= 1025 * 1025) return;
    int i = idx / 1025, j = idx % 1025;
    out[idx] = C[idx] + u[i] + v[j] - NORMC;
}

// ---------------------------------------------------------------------------
// Host side
// ---------------------------------------------------------------------------
static inline void launch_gemm(hipStream_t st,
                               const float* W, int wrm, int wrk,
                               const float* X, int xrk,
                               const float* X2, int x2rk, int ksplit,
                               int Cout, int Cin, int Ncol,
                               const float* bias, const float* bns, const float* bnb,
                               int relu, const float* cscale,
                               const float* resid, int rrm,
                               float alpha, float* Y, int yrm)
{
    dim3 grid(Ncol / 64, Cout / 16);
    if (!X2) { X2 = X; x2rk = xrk; }
    gemm_wmma_kernel<<<grid, dim3(128), 0, st>>>(W, wrm, wrk, X, xrk, X2, x2rk, ksplit,
                                                 Cin, bias, bns, bnb, relu, cscale,
                                                 resid, rrm, alpha, Y, yrm);
}

extern "C" void kernel_launch(void* const* d_in, const int* in_sizes, int n_in,
                              void* d_out, int out_size, void* d_ws, size_t ws_size,
                              hipStream_t stream)
{
    (void)in_sizes; (void)n_in; (void)out_size; (void)ws_size;

    const float* pf       = (const float*)d_in[0];
    const float* kpts0    = (const float*)d_in[1];
    const float* scores0  = (const float*)d_in[2];
    const float* desc1_in = (const float*)d_in[3];
    const float* kpts1    = (const float*)d_in[4];
    const float* scores1  = (const float*)d_in[5];
    const float* enc_w1   = (const float*)d_in[6];
    const float* enc_b1   = (const float*)d_in[7];
    const float* enc_bns  = (const float*)d_in[8];
    const float* enc_bnb  = (const float*)d_in[9];
    const float* enc_w2   = (const float*)d_in[10];
    const float* enc_b2   = (const float*)d_in[11];
    const float* kw1      = (const float*)d_in[12];
    const float* kb1      = (const float*)d_in[13];
    const float* kbn1s    = (const float*)d_in[14];
    const float* kbn1b    = (const float*)d_in[15];
    const float* kw2      = (const float*)d_in[16];
    const float* kb2      = (const float*)d_in[17];
    const float* kbn2s    = (const float*)d_in[18];
    const float* kbn2b    = (const float*)d_in[19];
    const float* kw3      = (const float*)d_in[20];
    const float* kb3      = (const float*)d_in[21];
    const float* proj_w   = (const float*)d_in[22];
    const float* proj_b   = (const float*)d_in[23];
    const float* merge_w  = (const float*)d_in[24];
    const float* merge_b  = (const float*)d_in[25];
    const float* mlp_w1   = (const float*)d_in[26];
    const float* mlp_b1   = (const float*)d_in[27];
    const float* mlp_bns  = (const float*)d_in[28];
    const float* mlp_bnb  = (const float*)d_in[29];
    const float* mlp_w2   = (const float*)d_in[30];
    const float* mlp_b2   = (const float*)d_in[31];
    const float* final_w  = (const float*)d_in[32];
    const float* final_b  = (const float*)d_in[33];
    const float* bin      = (const float*)d_in[34];

    float* out = (float*)d_out;

    // ---- workspace carve (floats, 256-aligned regions) ----
    float* wsp = (float*)d_ws;
    size_t off = 0;
    auto take = [&](size_t nfl) { float* p = wsp + off; off += (nfl + 255) & ~(size_t)255; return p; };
    float* dA0  = take(64 * 1024);
    float* dB0  = take(64 * 1024);
    float* dA1  = take(64 * 1024);
    float* dB1  = take(64 * 1024);
    float* qkv0 = take(192 * 1024);
    float* qkv1 = take(192 * 1024);
    float* S    = take(1024 * 1024);
    float* rinv = take(1024);
    float* msg  = take(64 * 1024);
    float* mrg  = take(64 * 1024);
    float* h2   = take(128 * 1024);
    float* md0  = take(64 * 1024);
    float* md1  = take(64 * 1024);
    float* kx   = take(5 * 1024);
    float* t32  = take(32 * 1024);
    float* t64  = take(64 * 1024);
    float* C    = take(1025 * 1025);
    float* uu   = take(1025);
    float* vv   = take(1025);

    // ---- descriptor init ----
    enc_kernel<<<1024, 64, 0, stream>>>(pf, enc_w1, enc_b1, enc_bns, enc_bnb,
                                        enc_w2, enc_b2, dA0);
    transpose_desc_kernel<<<256, 256, 0, stream>>>(desc1_in, dB0);

    auto kenc = [&](const float* kpts, const float* scores, float* desc) {
        prep_kx_kernel<<<20, 256, 0, stream>>>(kpts, scores, kx);
        launch_gemm(stream, kw1, 5, 1, kx, 1024, nullptr, 0, BIGSPLIT,
                    32, 5, 1024, kb1, kbn1s, kbn1b, 1, nullptr, nullptr, 0, 1.f, t32, 1024);
        launch_gemm(stream, kw2, 32, 1, t32, 1024, nullptr, 0, BIGSPLIT,
                    64, 32, 1024, kb2, kbn2s, kbn2b, 1, nullptr, nullptr, 0, 1.f, t64, 1024);
        launch_gemm(stream, kw3, 64, 1, t64, 1024, nullptr, 0, BIGSPLIT,
                    64, 64, 1024, kb3, nullptr, nullptr, 0, nullptr, desc, 1024, 1.f, desc, 1024);
    };
    kenc(kpts0, scores0, dA0);
    kenc(kpts1, scores1, dB0);

    float* curA = dA0; float* curB = dB0;
    float* nxtA = dA1; float* nxtB = dB1;

    // ---- one attentional branch: x -> x + mlp(concat(x, merge(attn))) ----
    auto branch = [&](const float* x, const float* qq, const float* kvsrc, float* o,
                      const float* mw, const float* mb,
                      const float* w1, const float* b1, const float* bs, const float* bb,
                      const float* w2, const float* b2) {
        const float* qp = qq;
        const float* kp = kvsrc + 64 * 1024;
        const float* vp = kvsrc + 128 * 1024;
        for (int h = 0; h < 4; ++h) {
            // S_T[m][n] = (1/4) * sum_d k_h[d][m] * q_h[d][n]
            launch_gemm(stream, kp + h * 1024, 1, 4096, qp + h * 1024, 4096,
                        nullptr, 0, BIGSPLIT, 1024, 16, 1024,
                        nullptr, nullptr, nullptr, 0, nullptr, nullptr, 0,
                        0.25f, S, 1024);
            softmax_col_kernel<<<4, 256, 0, stream>>>(S, rinv);
            // msg_h[d][n] = (sum_m v_h[d][m] * exp_S_T[m][n]) * rinv[n]
            launch_gemm(stream, vp + h * 1024, 4096, 1, S, 1024,
                        nullptr, 0, BIGSPLIT, 16, 1024, 1024,
                        nullptr, nullptr, nullptr, 0, rinv, nullptr, 0,
                        1.f, msg + h * 1024, 4096);
        }
        launch_gemm(stream, mw, 64, 1, msg, 1024, nullptr, 0, BIGSPLIT,
                    64, 64, 1024, mb, nullptr, nullptr, 0, nullptr, nullptr, 0, 1.f, mrg, 1024);
        // h2 = relu(bn(W1 @ [x; mrg] + b1))   (split-K input: rows 0..63 from x, 64..127 from mrg)
        launch_gemm(stream, w1, 128, 1, x, 1024, mrg, 1024, 64,
                    128, 128, 1024, b1, bs, bb, 1, nullptr, nullptr, 0, 1.f, h2, 1024);
        // o = x + W2 @ h2 + b2
        launch_gemm(stream, w2, 128, 1, h2, 1024, nullptr, 0, BIGSPLIT,
                    64, 128, 1024, b2, nullptr, nullptr, 0, nullptr, x, 1024, 1.f, o, 1024);
    };

    for (int i = 0; i < 18; ++i) {
        const float* pw = proj_w  + (size_t)i * 3 * 64 * 64;   // [192][64]
        const float* pb = proj_b  + (size_t)i * 192;
        const float* mw = merge_w + (size_t)i * 64 * 64;
        const float* mb = merge_b + (size_t)i * 64;
        const float* w1 = mlp_w1  + (size_t)i * 128 * 128;
        const float* b1 = mlp_b1  + (size_t)i * 128;
        const float* bs = mlp_bns + (size_t)i * 128;
        const float* bb = mlp_bnb + (size_t)i * 128;
        const float* w2 = mlp_w2  + (size_t)i * 64 * 128;
        const float* b2 = mlp_b2  + (size_t)i * 64;

        // Full q|k|v projection of both descriptor sets (covers self & cross layers)
        launch_gemm(stream, pw, 64, 1, curA, 1024, nullptr, 0, BIGSPLIT,
                    192, 64, 1024, pb, nullptr, nullptr, 0, nullptr, nullptr, 0, 1.f, qkv0, 1024);
        launch_gemm(stream, pw, 64, 1, curB, 1024, nullptr, 0, BIGSPLIT,
                    192, 64, 1024, pb, nullptr, nullptr, 0, nullptr, nullptr, 0, 1.f, qkv1, 1024);

        const float* src0 = (i % 2 == 0) ? qkv0 : qkv1;  // k/v source for branch 0
        const float* src1 = (i % 2 == 0) ? qkv1 : qkv0;

        branch(curA, qkv0, src0, nxtA, mw, mb, w1, b1, bs, bb, w2, b2);
        branch(curB, qkv1, src1, nxtB, mw, mb, w1, b1, bs, bb, w2, b2);

        float* t;
        t = curA; curA = nxtA; nxtA = t;
        t = curB; curB = nxtB; nxtB = t;
    }

    // ---- final projection + score matrix into couplings interior ----
    launch_gemm(stream, final_w, 64, 1, curA, 1024, nullptr, 0, BIGSPLIT,
                64, 64, 1024, final_b, nullptr, nullptr, 0, nullptr, nullptr, 0, 1.f, md0, 1024);
    launch_gemm(stream, final_w, 64, 1, curB, 1024, nullptr, 0, BIGSPLIT,
                64, 64, 1024, final_b, nullptr, nullptr, 0, nullptr, nullptr, 0, 1.f, md1, 1024);
    // C[n][m] = (1/8) * sum_d md0[d][n] * md1[d][m]   (W strided as md0^T)
    launch_gemm(stream, md0, 1, 1024, md1, 1024, nullptr, 0, BIGSPLIT,
                1024, 64, 1024, nullptr, nullptr, nullptr, 0, nullptr, nullptr, 0,
                0.125f, C, 1025);
    fill_bins_kernel<<<5, 256, 0, stream>>>(C, bin);

    // ---- Sinkhorn (100 iterations) ----
    init_uv_kernel<<<9, 256, 0, stream>>>(uu, vv);
    for (int it = 0; it < 100; ++it) {
        sink_row_kernel<<<1025, 256, 0, stream>>>(C, vv, uu);
        sink_col_kernel<<<5, 256, 0, stream>>>(C, uu, vv);
    }
    final_out_kernel<<<(1025 * 1025 + 255) / 256, 256, 0, stream>>>(C, uu, vv, out);
}
